// SelfAttention2D_31061203484908
// MI455X (gfx1250) — matrix operand order repaired and checked by reading, not compiled, not run on hardware
//
#include <hip/hip_runtime.h>
#include <hip/hip_bf16.h>

// ---------------------------------------------------------------------------
// Self-attention (B=2, S=2048, D=1024, H=16, DK=64) for gfx1250 (MI455X).
// All matrix math goes through v_wmma_f32_16x16x32_bf16 (fp32 accumulate).
// Workspace layout (bytes):
//   [ 0,  8M)  xb   : x as bf16            [B,S,D]
//   [ 8M,16M)  wq/wk/wv/wo bf16 (2MB each) [D,D]
//   [16M,24M)  Q    : bf16 [B,H,S,DK]
//   [24M,32M)  K    : bf16 [B,H,S,DK]
//   [32M,40M)  Vt   : bf16 [B,H,DK,S]   (transposed for p@v B-fragments)
//   [40M,48M)  ctx  : bf16 [B,S,D]      (merged-head attention output)
// ---------------------------------------------------------------------------

typedef __bf16 bf16;
typedef bf16  v16bf __attribute__((ext_vector_type(16)));
typedef bf16  v8bf  __attribute__((ext_vector_type(8)));
typedef float v8f   __attribute__((ext_vector_type(8)));

#define CB   2
#define CS   2048
#define CD   1024
#define CH   16
#define CDK  64

// ---- WMMA wrapper ---------------------------------------------------------
__device__ __forceinline__ v8f wmma_bf16(v16bf a, v16bf b, v8f c) {
  // (neg_a, A, neg_b, B, c_mod, C, reuse_a, reuse_b)
  return __builtin_amdgcn_wmma_f32_16x16x32_bf16(false, a, false, b,
                                                 (short)0, c, false, false);
}

// ---- DPP16 butterfly reduction over each 16-lane half (pure VALU) ---------
template <int CTRL>
__device__ __forceinline__ float dpp_mov(float x) {
  int xi = __builtin_bit_cast(int, x);
  int r  = __builtin_amdgcn_update_dpp(xi, xi, CTRL, 0xF, 0xF, true);
  return __builtin_bit_cast(float, r);
}
// After xor1+xor2 the quads are uniform, so row_half_mirror == xor4; after
// that the 8-groups are uniform, so row_mirror == xor8.  DPP16 rows are the
// 16-lane halves of a wave32, so the two halves reduce independently.
__device__ __forceinline__ float red_max16(float v) {
  v = fmaxf(v, dpp_mov<0xB1>(v));    // quad_perm [1,0,3,2]
  v = fmaxf(v, dpp_mov<0x4E>(v));    // quad_perm [2,3,0,1]
  v = fmaxf(v, dpp_mov<0x141>(v));   // row_half_mirror
  v = fmaxf(v, dpp_mov<0x140>(v));   // row_mirror
  return v;
}
__device__ __forceinline__ float red_sum16(float v) {
  v += dpp_mov<0xB1>(v);
  v += dpp_mov<0x4E>(v);
  v += dpp_mov<0x141>(v);
  v += dpp_mov<0x140>(v);
  return v;
}

// A-fragment: 16x32 (MxK). lane<16: row l, K {k0..k0+7, k0+16..k0+23};
// lane>=16: row l, K {k0+8..k0+15, k0+24..k0+31}. Two 16B contiguous loads.
__device__ __forceinline__ v16bf load_a_frag(const bf16* __restrict__ base,
                                             int ld, int row0, int k0,
                                             int l, int half) {
  const bf16* p = base + (size_t)(row0 + l) * ld + (k0 + half * 8);
  v8bf lo = *(const v8bf*)p;
  v8bf hi = *(const v8bf*)(p + 16);
  return __builtin_shufflevector(lo, hi, 0,1,2,3,4,5,6,7,
                                         8,9,10,11,12,13,14,15);
}

// B-fragment: 32x16 (KxN) from a row-major [N, ldK] matrix (B[k][n] = M[n][k]).
// lane<16: col l, K {k0..k0+7, k0+16..k0+23}; lane>=16: col l, K {k0+8..k0+15, k0+24..k0+31}.
// Two 16B contiguous loads per lane.
__device__ __forceinline__ v16bf load_b_frag(const bf16* __restrict__ base,
                                             int ld, int col0, int k0,
                                             int l, int half) {
  const bf16* p = base + (size_t)(col0 + l) * ld + (k0 + half * 8);
  v8bf lo = *(const v8bf*)p;
  v8bf hi = *(const v8bf*)(p + 16);
  return __builtin_shufflevector(lo, hi, 0,1,2,3,4,5,6,7,
                                         8,9,10,11,12,13,14,15);
}

// ---- fp32 -> bf16 conversion ---------------------------------------------
__global__ __launch_bounds__(256) void cvt_f32_bf16(const float* __restrict__ in,
                                                    bf16* __restrict__ out, int n) {
  int i = blockIdx.x * 256 + threadIdx.x;
  if (i < n) out[i] = (bf16)in[i];
}

// ---- Generic WMMA GEMM: C[M=4096, N=1024] = A[M,K=1024] @ W[N,K]^T + bias --
// MODE 0: out bf16 head-split [B,H,S,DK]
// MODE 1: out bf16 transposed [B,H,DK,S]   (for V; vectorized 16B stores)
// MODE 2: out fp32 plain [M,N]             (final projection -> d_out)
template <int MODE>
__global__ __launch_bounds__(256) void gemm_wmma(const bf16* __restrict__ A,
                                                 const bf16* __restrict__ W,
                                                 const float* __restrict__ bias,
                                                 void* __restrict__ outp) {
  const int K = CD, N = CD;
  const int lane = threadIdx.x & 31;
  const int l = lane & 15, half = lane >> 4;
  const int w = threadIdx.x >> 5;       // 8 waves
  const int wm = w >> 1, wn = w & 1;    // 4x2 wave grid -> block tile 128x128
  const int mBase = blockIdx.y * 128 + wm * 32;
  const int nBase = blockIdx.x * 128 + wn * 64;

  v8f acc[2][4];
#pragma unroll
  for (int mi = 0; mi < 2; ++mi)
#pragma unroll
    for (int t = 0; t < 4; ++t) acc[mi][t] = (v8f){};

  for (int k0 = 0; k0 < K; k0 += 32) {
    v16bf a0 = load_a_frag(A, K, mBase,      k0, l, half);
    v16bf a1 = load_a_frag(A, K, mBase + 16, k0, l, half);
#pragma unroll
    for (int t = 0; t < 4; ++t) {
      v16bf bt = load_b_frag(W, K, nBase + t * 16, k0, l, half);
      acc[0][t] = wmma_bf16(a0, bt, acc[0][t]);
      acc[1][t] = wmma_bf16(a1, bt, acc[1][t]);
    }
  }

#pragma unroll
  for (int mi = 0; mi < 2; ++mi) {
#pragma unroll
    for (int t = 0; t < 4; ++t) {
      const int n = nBase + t * 16 + l;
      const float bv = bias[n];
      if (MODE == 1) {
        // j indexes consecutive s -> one contiguous v8bf store per accumulator
        const int m0 = mBase + mi * 16 + half * 8;     // first row of this half
        const int b = m0 >> 11, s0 = m0 & (CS - 1);
        const int h = n >> 6,   dk = n & (CDK - 1);
        v8bf pk;
#pragma unroll
        for (int j = 0; j < 8; ++j) pk[j] = (bf16)(acc[mi][t][j] + bv);
        const size_t idx = ((size_t)((b * CH + h) * CDK + dk)) * CS + s0;
        *(v8bf*)((bf16*)outp + idx) = pk;
      } else {
#pragma unroll
        for (int j = 0; j < 8; ++j) {
          const int m = mBase + mi * 16 + half * 8 + j;  // C-layout row
          const float val = acc[mi][t][j] + bv;
          if (MODE == 2) {
            ((float*)outp)[(size_t)m * N + n] = val;
          } else {
            const int b = m >> 11, s = m & (CS - 1);
            const int h = n >> 6,  dk = n & (CDK - 1);
            const size_t idx = ((size_t)((b * CH + h) * CS + s)) * CDK + dk;
            ((bf16*)outp)[idx] = (bf16)val;
          }
        }
      }
    }
  }
}

// ---- Flash attention ------------------------------------------------------
// grid = (S/64, B*H), block = 128 (4 waves). Wave owns 16 query rows.
__global__ __launch_bounds__(128) void flash_attn(const bf16* __restrict__ Q,
                                                  const bf16* __restrict__ Km,
                                                  const bf16* __restrict__ Vt,
                                                  const int* __restrict__ amask,
                                                  bf16* __restrict__ ctx) {
  const int lane = threadIdx.x & 31;
  const int l = lane & 15, half = lane >> 4;
  const int w = threadIdx.x >> 5;
  const int bh = blockIdx.y;
  const int b = bh >> 4, h = bh & 15;
  const int q0 = blockIdx.x * 64 + w * 16;

  const bf16* Qh = Q  + (size_t)bh * CS * CDK;
  const bf16* Kh = Km + (size_t)bh * CS * CDK;
  const bf16* Vh = Vt + (size_t)bh * CDK * CS;
  const int* mrow = amask + b * CS;

  // per-wave probability tile: 16 rows x 32 keys, row stride 40 (80B, 16B-aligned)
  __shared__ bf16 P[4][16][40];

  const v16bf qf0 = load_a_frag(Qh, CDK, q0, 0,  l, half);
  const v16bf qf1 = load_a_frag(Qh, CDK, q0, 32, l, half);

  v8f o[4];
#pragma unroll
  for (int a = 0; a < 4; ++a) o[a] = (v8f){};
  float rm[8], rl[8];
#pragma unroll
  for (int j = 0; j < 8; ++j) { rm[j] = -3.4028235e38f; rl[j] = 0.0f; }

  for (int kv = 0; kv < CS; kv += 32) {
    // scores: 16 q-rows x 32 keys  (DK=64 -> 2 K-chunks per 16-key tile)
    v8f s0 = (v8f){}, s1 = (v8f){};
    s0 = wmma_bf16(qf0, load_b_frag(Kh, CDK, kv,      0,  l, half), s0);
    s0 = wmma_bf16(qf1, load_b_frag(Kh, CDK, kv,      32, l, half), s0);
    s1 = wmma_bf16(qf0, load_b_frag(Kh, CDK, kv + 16, 0,  l, half), s1);
    s1 = wmma_bf16(qf1, load_b_frag(Kh, CDK, kv + 16, 32, l, half), s1);

    const bool keep0 = mrow[kv + l] != 0;
    const bool keep1 = mrow[kv + 16 + l] != 0;

#pragma unroll
    for (int j = 0; j < 8; ++j) {
      // reference: sim / DK**-0.5  ==  sim * sqrt(DK) = sim * 8
      float v0 = keep0 ? s0[j] * 8.0f : -3.4028235e38f;
      float v1 = keep1 ? s1[j] * 8.0f : -3.4028235e38f;
      const float tm    = red_max16(fmaxf(v0, v1));   // row max (per half)
      const float newm  = fmaxf(rm[j], tm);
      const float alpha = __expf(rm[j] - newm);
      const float p0 = keep0 ? __expf(v0 - newm) : 0.0f;
      const float p1 = keep1 ? __expf(v1 - newm) : 0.0f;
      const float ps = red_sum16(p0 + p1);            // row sum (per half)
      rl[j] = rl[j] * alpha + ps;
      rm[j] = newm;
#pragma unroll
      for (int a = 0; a < 4; ++a) o[a][j] *= alpha;
      const int r = half * 8 + j;                     // C-layout row -> tile row
      P[w][r][l]      = (bf16)p0;
      P[w][r][l + 16] = (bf16)p1;
    }

    // cross-lane LDS write->read hazard inside the wave: drain DS counter
    asm volatile("s_wait_dscnt 0" ::: "memory");

    // re-shape C-layout -> A-layout: lane row l, keys {half*8.., +16..}
    const bf16* pp = &P[w][l][half * 8];
    v8bf plo = *(const v8bf*)pp;
    v8bf phi = *(const v8bf*)(pp + 16);
    v16bf pf = __builtin_shufflevector(plo, phi, 0,1,2,3,4,5,6,7,
                                                 8,9,10,11,12,13,14,15);

#pragma unroll
    for (int a = 0; a < 4; ++a) {
      // B[k=key][n=dk] = Vt[dk][key]  -> contiguous along keys
      v16bf vb = load_b_frag(Vh, CS, a * 16, kv, l, half);
      o[a] = wmma_bf16(pf, vb, o[a]);
    }
  }

  // normalize and write merged-head ctx [B,S,D] (bf16)
#pragma unroll
  for (int j = 0; j < 8; ++j) {
    const float inv = 1.0f / rl[j];
    const int s = q0 + half * 8 + j;
#pragma unroll
    for (int a = 0; a < 4; ++a) {
      const int d = h * CDK + a * 16 + l;
      ctx[((size_t)(b * CS + s)) * CD + d] = (bf16)(o[a][j] * inv);
    }
  }
}

// ---------------------------------------------------------------------------
extern "C" void kernel_launch(void* const* d_in, const int* in_sizes, int n_in,
                              void* d_out, int out_size, void* d_ws, size_t ws_size,
                              hipStream_t stream) {
  const float* x     = (const float*)d_in[0];
  const int*   amask = (const int*)  d_in[1];
  const float* Wq    = (const float*)d_in[2];
  const float* bq    = (const float*)d_in[3];
  const float* Wk    = (const float*)d_in[4];
  const float* bk    = (const float*)d_in[5];
  const float* Wv    = (const float*)d_in[6];
  const float* bv    = (const float*)d_in[7];
  const float* Wo    = (const float*)d_in[8];
  const float* bo    = (const float*)d_in[9];

  char* ws = (char*)d_ws;
  bf16* xb  = (bf16*)(ws);                       // 8 MB
  bf16* wqb = (bf16*)(ws + (8u  << 20));         // 2 MB each
  bf16* wkb = wqb + (1u << 20);
  bf16* wvb = wkb + (1u << 20);
  bf16* wob = wvb + (1u << 20);
  bf16* Qh  = (bf16*)(ws + (16u << 20));         // [B,H,S,DK]
  bf16* Kh  = (bf16*)(ws + (24u << 20));         // [B,H,S,DK]
  bf16* Vt  = (bf16*)(ws + (32u << 20));         // [B,H,DK,S]
  bf16* ctx = (bf16*)(ws + (40u << 20));         // [B,S,D]

  const int nX = CB * CS * CD;                   // 4,194,304
  const int nW = CD * CD;                        // 1,048,576
  cvt_f32_bf16<<<(nX + 255) / 256, 256, 0, stream>>>(x,  xb,  nX);
  cvt_f32_bf16<<<(nW + 255) / 256, 256, 0, stream>>>(Wq, wqb, nW);
  cvt_f32_bf16<<<(nW + 255) / 256, 256, 0, stream>>>(Wk, wkb, nW);
  cvt_f32_bf16<<<(nW + 255) / 256, 256, 0, stream>>>(Wv, wvb, nW);
  cvt_f32_bf16<<<(nW + 255) / 256, 256, 0, stream>>>(Wo, wob, nW);

  dim3 gg(CD / 128, (CB * CS) / 128);            // (8, 32)
  gemm_wmma<0><<<gg, 256, 0, stream>>>(xb, wqb, bq, (void*)Qh);
  gemm_wmma<0><<<gg, 256, 0, stream>>>(xb, wkb, bk, (void*)Kh);
  gemm_wmma<1><<<gg, 256, 0, stream>>>(xb, wvb, bv, (void*)Vt);

  flash_attn<<<dim3(CS / 64, CB * CH), 128, 0, stream>>>(Qh, Kh, Vt, amask, ctx);

  gemm_wmma<2><<<gg, 256, 0, stream>>>(ctx, wob, bo, d_out);
}
